// LIA_19653770346889
// MI455X (gfx1250) — matrix operand order repaired and checked by reading, not compiled, not run on hardware
//
#include <hip/hip_runtime.h>
#include <hip/hip_bf16.h>
#include <math.h>

typedef __attribute__((ext_vector_type(16))) _Float16 v16h;
typedef __attribute__((ext_vector_type(8)))  float    v8f;
typedef __attribute__((ext_vector_type(2)))  float    v2f;

#define KKTOT 100000      // sampled antenna positions
#define NSAR  512
#define BA    2000        // A*B image pixels
#define BAP   2048        // zero-padded column count
#define AA    50
#define BB    40
#define KT    32          // K per f16-WMMA macro step; 100000 = 3125*32
#define NKSTEP (KKTOT / KT)
#define NGRP  (BAP / 16)  // 128 column groups of 16
#define NB    40          // elimination block size; 2000 = 50*40
#define DI_F  0.01f

__device__ __forceinline__ float kconst() {
  return (float)(2.0 * 3.14159265358979323846 * 77.0e9 * 2.0 / 299792458.0);
}

// Fragment-block linear index (in v16h units): plane[(kt*NGRP + g)*32 + lane]
__device__ __forceinline__ size_t fragIdx(int kt, int g, int lane) {
  return ((size_t)kt * NGRP + (size_t)g) * 32 + (size_t)lane;
}

// ---------------------------------------------------------------------------
// 1) Build Hp phasors directly in WMMA fragment layouts (global memory).
//    Planes: A-layout cos / +sin / -sin ; B-layout cos / +sin.
//    Each thread computes 4 octets of 8 consecutive kk for one column j -> one aligned
//    16B store per plane per octet (A and B octet payloads are identical).
// ---------------------------------------------------------------------------
__global__ void __launch_bounds__(256)
build_hp(const int* __restrict__ py,
         _Float16* __restrict__ aRe, _Float16* __restrict__ aIp,
         _Float16* __restrict__ aIn,
         _Float16* __restrict__ bRe, _Float16* __restrict__ bIp) {
  const int kt = blockIdx.x;            // K-tile (32 rows)
  const int j  = blockIdx.y * 256 + threadIdx.x;  // 0..BAP-1
  const int g  = j >> 4;
  const int m  = j & 15;

  __shared__ int s_py[KT];
  if (threadIdx.x < KT) s_py[threadIdx.x] = py[kt * KT + threadIdx.x];
  __syncthreads();

  const float KC = kconst();
  const float z2 = 0.09f;
  float wy = 0.0f, wx = 0.0f;
  if (j < BA) {
    const int jy = j % BB;
    const int jx = j / BB;
    wy = (-100.0f + (float)jy * (200.0f / 39.0f)) * 1.0e-3f;
    wx = (-100.0f + (float)jx * (200.0f / 49.0f)) * 1.0e-3f;
  }

  union Pack { _Float16 h[8]; uint4 u; };

#pragma unroll
  for (int o = 0; o < 4; ++o) {         // octet of kk values
    Pack pc, ps, pn;
#pragma unroll
    for (int s = 0; s < 8; ++s) {
      float c = 0.0f, si = 0.0f;
      if (j < BA) {
        const int p = s_py[o * 8 + s];
        const float ay = ((float)(p % NSAR) + 0.5f - 256.0f) * 1.0e-3f;
        const float ax = ((float)(p / NSAR) + 0.5f - 256.0f) * 1.0e-3f;
        const float dy = ay - wy, dx = ax - wx;
        const float dist = sqrtf(dy * dy + dx * dx + z2);
        __sincosf(KC * dist, &si, &c);
      }
      pc.h[s] = (_Float16)c;
      ps.h[s] = (_Float16)si;
      pn.h[s] = (_Float16)(-si);
    }
    // A layout: lane = m + 16*(o>>1), elem base = 8*(o&1)
    const size_t dA = fragIdx(kt, g, m + 16 * (o >> 1)) * 16 + 8 * (o & 1);
    // B layout: lane = m + 16*(o>>1), elem base = 8*(o&1)
    const size_t dB = fragIdx(kt, g, m + 16 * (o >> 1)) * 16 + 8 * (o & 1);
    *(uint4*)&aRe[dA] = pc.u;
    *(uint4*)&aIp[dA] = ps.u;
    *(uint4*)&aIn[dA] = pn.u;
    *(uint4*)&bRe[dB] = pc.u;
    *(uint4*)&bIp[dB] = ps.u;
  }
}

// ---------------------------------------------------------------------------
// 2) G = DI * Hp^H Hp, exploiting Hermitian symmetry (half the blocks).
//    Square 64x64 block tile: 4 waves (2x2) of 32x32 wave tiles, 128 threads.
//    Blocks strictly below the diagonal are skipped; strictly-upper blocks
//    also mirror-write G[c][r] = conj(G[r][c]) (unique writer -> no races).
//    No LDS: fragments read from pre-swizzled global planes (b128 clauses).
//    re = ar*br + ai*bi ; im = ar*bi + (-ai)*br  (all add-accumulate).
// ---------------------------------------------------------------------------
__global__ void __launch_bounds__(128)
gemm_g(const _Float16* __restrict__ aReP, const _Float16* __restrict__ aIpP,
       const _Float16* __restrict__ aInP,
       const _Float16* __restrict__ bReP, const _Float16* __restrict__ bIpP,
       float* __restrict__ gRe, float* __restrict__ gIm) {
  if (blockIdx.x > blockIdx.y) return;          // lower triangle: mirrored
  const bool mirror = (blockIdx.x < blockIdx.y);

  const int lane = threadIdx.x & 31;
  const int w    = threadIdx.x >> 5;            // 0..3
  const int wi   = w & 1;                       // wave row
  const int wj   = w >> 1;                      // wave col
  const int gA0  = blockIdx.x * 4 + wi * 2;     // row groups gA0, gA0+1
  const int gB0  = blockIdx.y * 4 + wj * 2;     // col groups gB0, gB0+1

  const v16h* Ar = (const v16h*)aReP;
  const v16h* Ai = (const v16h*)aIpP;
  const v16h* An = (const v16h*)aInP;
  const v16h* Br = (const v16h*)bReP;
  const v16h* Bi = (const v16h*)bIpP;

  const size_t iA0 = (size_t)gA0 * 32 + lane;
  const size_t iA1 = iA0 + 32;
  const size_t iB0 = (size_t)gB0 * 32 + lane;
  const size_t iB1 = iB0 + 32;

  v8f re00 = {}, re01 = {}, re10 = {}, re11 = {};
  v8f im00 = {}, im01 = {}, im10 = {}, im11 = {};

  size_t kbase = 0;                     // advances by NGRP*32 v16h per K-step
#pragma unroll 1
  for (int kt = 0; kt < NKSTEP; ++kt, kbase += (size_t)NGRP * 32) {
    const v16h ar0 = Ar[kbase + iA0], ar1 = Ar[kbase + iA1];
    const v16h ai0 = Ai[kbase + iA0], ai1 = Ai[kbase + iA1];
    const v16h an0 = An[kbase + iA0], an1 = An[kbase + iA1];
    const v16h br0 = Br[kbase + iB0], br1 = Br[kbase + iB1];
    const v16h bi0 = Bi[kbase + iB0], bi1 = Bi[kbase + iB1];

    re00 = __builtin_amdgcn_wmma_f32_16x16x32_f16(false, ar0, false, br0, (short)0, re00, false, false);
    re00 = __builtin_amdgcn_wmma_f32_16x16x32_f16(false, ai0, false, bi0, (short)0, re00, false, false);
    im00 = __builtin_amdgcn_wmma_f32_16x16x32_f16(false, ar0, false, bi0, (short)0, im00, false, false);
    im00 = __builtin_amdgcn_wmma_f32_16x16x32_f16(false, an0, false, br0, (short)0, im00, false, false);

    re01 = __builtin_amdgcn_wmma_f32_16x16x32_f16(false, ar0, false, br1, (short)0, re01, false, false);
    re01 = __builtin_amdgcn_wmma_f32_16x16x32_f16(false, ai0, false, bi1, (short)0, re01, false, false);
    im01 = __builtin_amdgcn_wmma_f32_16x16x32_f16(false, ar0, false, bi1, (short)0, im01, false, false);
    im01 = __builtin_amdgcn_wmma_f32_16x16x32_f16(false, an0, false, br1, (short)0, im01, false, false);

    re10 = __builtin_amdgcn_wmma_f32_16x16x32_f16(false, ar1, false, br0, (short)0, re10, false, false);
    re10 = __builtin_amdgcn_wmma_f32_16x16x32_f16(false, ai1, false, bi0, (short)0, re10, false, false);
    im10 = __builtin_amdgcn_wmma_f32_16x16x32_f16(false, ar1, false, bi0, (short)0, im10, false, false);
    im10 = __builtin_amdgcn_wmma_f32_16x16x32_f16(false, an1, false, br0, (short)0, im10, false, false);

    re11 = __builtin_amdgcn_wmma_f32_16x16x32_f16(false, ar1, false, br1, (short)0, re11, false, false);
    re11 = __builtin_amdgcn_wmma_f32_16x16x32_f16(false, ai1, false, bi1, (short)0, re11, false, false);
    im11 = __builtin_amdgcn_wmma_f32_16x16x32_f16(false, ar1, false, bi1, (short)0, im11, false, false);
    im11 = __builtin_amdgcn_wmma_f32_16x16x32_f16(false, an1, false, br1, (short)0, im11, false, false);
  }

  // C/D layout: element e -> row = base + e + 8*(lane>=16); col = base + (lane&15)
  const int rB = blockIdx.x * 64 + wi * 32 + ((lane >> 4) << 3);
  const int cB = blockIdx.y * 64 + wj * 32 + (lane & 15);
  const v8f* res[4] = { &re00, &re01, &re10, &re11 };
  const v8f* ims[4] = { &im00, &im01, &im10, &im11 };
#pragma unroll
  for (int t = 0; t < 4; ++t) {
    const int r0 = rB + (t >> 1) * 16;
    const int c  = cB + (t & 1) * 16;
#pragma unroll
    for (int e = 0; e < 8; ++e) {
      const int r = r0 + e;
      const float vr = DI_F * (*res[t])[e];
      const float vi = DI_F * (*ims[t])[e];
      gRe[(size_t)r * BAP + c] = vr;
      gIm[(size_t)r * BAP + c] = vi;
      if (mirror) {                      // Hermitian mirror, unique writer
        gRe[(size_t)c * BAP + r] = vr;
        gIm[(size_t)c * BAP + r] = -vi;
      }
    }
  }
}

// ---------------------------------------------------------------------------
// 3) xd = DI * Hp^H rd  (reads the B-layout planes; one block per column j).
// ---------------------------------------------------------------------------
__global__ void xd_kernel(const float* __restrict__ sar, const int* __restrict__ py,
                          const _Float16* __restrict__ bRe, const _Float16* __restrict__ bIp,
                          float* __restrict__ xdRe, float* __restrict__ xdIm) {
  const int j = blockIdx.x;
  const int g = j >> 4, m = j & 15;
  __shared__ float sRe[256], sIm[256];
  float ar = 0.0f, ai = 0.0f;
  if (j < BA) {
    for (int k = threadIdx.x; k < KKTOT; k += 256) {
      const int kt = k >> 5, kk = k & 31;
      const size_t off = fragIdx(kt, g, m + ((kk >> 4) << 4)) * 16 + (kk & 15);
      const float rdk = sar[py[k]];
      ar += (float)bRe[off] * rdk;
      ai -= (float)bIp[off] * rdk;      // conj
    }
  }
  sRe[threadIdx.x] = ar; sIm[threadIdx.x] = ai;
  __syncthreads();
  for (int s = 128; s > 0; s >>= 1) {
    if (threadIdx.x < (unsigned)s) {
      sRe[threadIdx.x] += sRe[threadIdx.x + s];
      sIm[threadIdx.x] += sIm[threadIdx.x + s];
    }
    __syncthreads();
  }
  if (threadIdx.x == 0) { xdRe[j] = DI_F * sRe[0]; xdIm[j] = DI_F * sIm[0]; }
}

// ---------------------------------------------------------------------------
// 4a) Panel factorization: NB sequential Gauss-Jordan steps confined to the
//     column/row panels. Single block; __syncthreads orders global traffic.
// ---------------------------------------------------------------------------
__global__ void __launch_bounds__(1024)
panel_factor(const float* __restrict__ gRe, const float* __restrict__ gIm,
             float* __restrict__ xdRe, float* __restrict__ xdIm,
             float* __restrict__ TR, float* __restrict__ TI,
             float* __restrict__ TnR, float* __restrict__ TnI,
             float* __restrict__ VtR, float* __restrict__ VtI,
             float* __restrict__ tcR, float* __restrict__ tcI, int j0) {
  __shared__ float sVkR[NB], sVkI[NB], sTkR[NB], sTkI[NB];
  __shared__ float sInvR, sInvI, sXkR, sXkI;
  const int tid = threadIdx.x;

  for (int m = 0; m < NB; ++m) {
    const int k = j0 + m;
    if (tid < m) { sVkR[tid] = VtR[(size_t)k * NB + tid]; sVkI[tid] = VtI[(size_t)k * NB + tid]; }
    __syncthreads();
    // current column k: original column minus partial T*V
    for (int r = tid; r < BAP; r += 1024) {
      float aR = gRe[(size_t)r * BAP + k];
      float aI = gIm[(size_t)r * BAP + k];
      for (int mp = 0; mp < m; ++mp) {
        const float tR = TR[(size_t)r * NB + mp], tI = TI[(size_t)r * NB + mp];
        aR -= tR * sVkR[mp] - tI * sVkI[mp];
        aI -= tR * sVkI[mp] + tI * sVkR[mp];
      }
      tcR[r] = aR; tcI[r] = aI;
    }
    __syncthreads();
    if (tid == 0) {
      const float cr = tcR[k], ci = tcI[k];
      const float dr = 1.0f + cr;
      const float den = dr * dr + ci * ci;
      sInvR = dr / den; sInvI = -ci / den;
      sXkR = xdRe[k]; sXkI = xdIm[k];
    }
    __syncthreads();
    for (int r = tid; r < BAP; r += 1024) {
      const float cr = tcR[r], ci = tcI[r];
      const float tR = cr * sInvR - ci * sInvI;
      const float tI = cr * sInvI + ci * sInvR;
      TR[(size_t)r * NB + m] = tR;  TI[(size_t)r * NB + m] = tI;
      TnR[(size_t)r * NB + m] = -tR; TnI[(size_t)r * NB + m] = -tI;
      xdRe[r] -= tR * sXkR - tI * sXkI;
      xdIm[r] -= tR * sXkI + tI * sXkR;
    }
    __syncthreads();
    if (tid < m) { sTkR[tid] = TR[(size_t)k * NB + tid]; sTkI[tid] = TI[(size_t)k * NB + tid]; }
    __syncthreads();
    // current row k: original row minus partial T*V; store transposed
    for (int c = tid; c < BAP; c += 1024) {
      float vR = gRe[(size_t)k * BAP + c];
      float vI = gIm[(size_t)k * BAP + c];
      for (int mp = 0; mp < m; ++mp) {
        const float uR = VtR[(size_t)c * NB + mp], uI = VtI[(size_t)c * NB + mp];
        vR -= sTkR[mp] * uR - sTkI[mp] * uI;
        vI -= sTkR[mp] * uI + sTkI[mp] * uR;
      }
      VtR[(size_t)c * NB + m] = vR; VtI[(size_t)c * NB + m] = vI;
    }
    __syncthreads();
  }
}

// ---------------------------------------------------------------------------
// 4b) Trailing rank-NB update: G -= T*V (complex) via v_wmma_f32_16x16x4_f32.
//     re: G + (-TR)*VR + TI*VI ;  im: G + (-TR)*VI + (-TI)*VR.
// ---------------------------------------------------------------------------
__global__ void __launch_bounds__(256)
rank_update(float* __restrict__ gRe, float* __restrict__ gIm,
            const float* __restrict__ TnR, const float* __restrict__ TI,
            const float* __restrict__ TnI,
            const float* __restrict__ VtR, const float* __restrict__ VtI) {
  const int lane = threadIdx.x & 31;
  const int w    = threadIdx.x >> 5;
  const int wi   = w & 3, wj = w >> 2;          // 4x2 waves -> block tile 64x32
  const int r0   = blockIdx.x * 64 + wi * 16;
  const int c0   = blockIdx.y * 32 + wj * 16;

  const int rEl = r0 + ((lane >> 4) << 3);      // C/D row base for element e
  const int cEl = c0 + (lane & 15);

  v8f accR, accI;
#pragma unroll
  for (int e = 0; e < 8; ++e) {
    accR[e] = gRe[(size_t)(rEl + e) * BAP + cEl];
    accI[e] = gIm[(size_t)(rEl + e) * BAP + cEl];
  }

  const size_t aRow = (size_t)(r0 + (lane & 15)) * NB + 2 * (lane >> 4);
  const size_t bRow = (size_t)(c0 + (lane & 15)) * NB + 2 * (lane >> 4);
#pragma unroll
  for (int kb = 0; kb < NB; kb += 4) {
    const v2f tnr = *(const v2f*)&TnR[aRow + kb];
    const v2f tpi = *(const v2f*)&TI [aRow + kb];
    const v2f tni = *(const v2f*)&TnI[aRow + kb];
    const v2f vr  = *(const v2f*)&VtR[bRow + kb];
    const v2f vi  = *(const v2f*)&VtI[bRow + kb];
    accR = __builtin_amdgcn_wmma_f32_16x16x4_f32(false, tnr, false, vr, (short)0, accR, false, false);
    accR = __builtin_amdgcn_wmma_f32_16x16x4_f32(false, tpi, false, vi, (short)0, accR, false, false);
    accI = __builtin_amdgcn_wmma_f32_16x16x4_f32(false, tnr, false, vi, (short)0, accI, false, false);
    accI = __builtin_amdgcn_wmma_f32_16x16x4_f32(false, tni, false, vr, (short)0, accI, false, false);
  }

#pragma unroll
  for (int e = 0; e < 8; ++e) {
    gRe[(size_t)(rEl + e) * BAP + cEl] = accR[e];
    gIm[(size_t)(rEl + e) * BAP + cEl] = accI[e];
  }
}

// ---------------------------------------------------------------------------
// 5) xd /= diag(G); reshape (B,A), flip axis=1; interleaved complex64 out.
// ---------------------------------------------------------------------------
__global__ void finalize(const float* __restrict__ gRe, const float* __restrict__ gIm,
                         const float* __restrict__ xdRe, const float* __restrict__ xdIm,
                         float* __restrict__ out) {
  const int idx = blockIdx.x * 256 + threadIdx.x;
  if (idx >= BA) return;
  const int r = idx / AA;
  const int c = idx % AA;
  const int q = r * AA + (AA - 1 - c);
  const float dr = gRe[(size_t)q * BAP + q];
  const float di = gIm[(size_t)q * BAP + q];
  const float den = dr * dr + di * di;
  const float xr = xdRe[q], xi = xdIm[q];
  out[idx * 2 + 0] = (xr * dr + xi * di) / den;
  out[idx * 2 + 1] = (xi * dr - xr * di) / den;
}

// ---------------------------------------------------------------------------
extern "C" void kernel_launch(void* const* d_in, const int* in_sizes, int n_in,
                              void* d_out, int out_size, void* d_ws, size_t ws_size,
                              hipStream_t stream) {
  const float* sar = (const float*)d_in[0];
  const int*   py  = (const int*)d_in[1];
  float* out = (float*)d_out;

  char* ws = (char*)d_ws;
  const size_t szPlane = (size_t)KKTOT * BAP * sizeof(_Float16); // 409.6 MB
  const size_t szG  = (size_t)BAP * BAP * sizeof(float);         // 16.8 MB
  const size_t szV  = (size_t)BAP * sizeof(float);
  const size_t szP  = (size_t)BAP * NB * sizeof(float);          // 327.7 KB
  size_t off = 0;
  _Float16* aRe = (_Float16*)(ws + off); off += szPlane;
  _Float16* aIp = (_Float16*)(ws + off); off += szPlane;
  _Float16* aIn = (_Float16*)(ws + off); off += szPlane;
  _Float16* bRe = (_Float16*)(ws + off); off += szPlane;
  _Float16* bIp = (_Float16*)(ws + off); off += szPlane;
  float* gRe = (float*)(ws + off); off += szG;
  float* gIm = (float*)(ws + off); off += szG;
  float* xdRe = (float*)(ws + off); off += szV;
  float* xdIm = (float*)(ws + off); off += szV;
  float* tcR = (float*)(ws + off); off += szV;
  float* tcI = (float*)(ws + off); off += szV;
  float* TR  = (float*)(ws + off); off += szP;
  float* TI  = (float*)(ws + off); off += szP;
  float* TnR = (float*)(ws + off); off += szP;
  float* TnI = (float*)(ws + off); off += szP;
  float* VtR = (float*)(ws + off); off += szP;
  float* VtI = (float*)(ws + off); off += szP;

  // 1) Hp phasors, pre-swizzled into WMMA fragment layouts
  build_hp<<<dim3(NKSTEP, BAP / 256), 256, 0, stream>>>(py, aRe, aIp, aIn, bRe, bIp);

  // 2) G = DI * Hp^H Hp  (f16 WMMA, Hermitian: upper-triangle blocks + mirror)
  gemm_g<<<dim3(BAP / 64, BAP / 64), 128, 0, stream>>>(aRe, aIp, aIn, bRe, bIp, gRe, gIm);

  // 3) xd = DI * Hp^H rd
  xd_kernel<<<BAP, 256, 0, stream>>>(sar, py, bRe, bIp, xdRe, xdIm);

  // 4) blocked Gauss-Jordan: 50 x (panel factor + rank-40 WMMA trailing update)
  for (int b = 0; b < BA / NB; ++b) {
    panel_factor<<<1, 1024, 0, stream>>>(gRe, gIm, xdRe, xdIm,
                                         TR, TI, TnR, TnI, VtR, VtI, tcR, tcI, b * NB);
    rank_update<<<dim3(BAP / 64, BAP / 32), 256, 0, stream>>>(gRe, gIm, TnR, TI, TnI, VtR, VtI);
  }

  // 5) divide by diagonal, flip, emit interleaved complex64
  finalize<<<(BA + 255) / 256, 256, 0, stream>>>(gRe, gIm, xdRe, xdIm, out);
}
